// MambaLayer_67980742361144
// MI455X (gfx1250) — matrix operand order repaired and checked by reading, not compiled, not run on hardware
//
#include <hip/hip_runtime.h>
#include <hip/hip_bf16.h>

typedef __attribute__((ext_vector_type(16))) _Float16 v16h;
typedef __attribute__((ext_vector_type(8)))  _Float16 v8h;
typedef __attribute__((ext_vector_type(8)))  float    v8f;

#define DIM_C    384
#define D_INNER  768
#define D_STATE  16
#define DT_RANK  24
#define L_SEQ    4096
#define B_SZ     4
#define BL_TOK   (B_SZ * L_SEQ)          // 16384 tokens
#define N_XZ     (2 * D_INNER)           // 1536
#define N_XDBL   (DT_RANK + 2 * D_STATE) // 56

// ---------------------------------------------------------------------------
// f32 -> f16 conversion (weights / activations)
// ---------------------------------------------------------------------------
__global__ void cvt_f32_f16(const float* __restrict__ src, _Float16* __restrict__ dst, int n) {
    int i = blockIdx.x * 256 + threadIdx.x;
    if (i < n) dst[i] = (_Float16)src[i];
}

// ---------------------------------------------------------------------------
// LayerNorm over C for each token; input x is (B, C, L), output xn is
// (B*L, C) in f16 (A-matrix layout for WMMA). 32 tokens per block staged
// through padded LDS so both the strided reads and the f16 writes coalesce.
// ---------------------------------------------------------------------------
__global__ __launch_bounds__(256)
void ln_kernel(const float* __restrict__ x, const float* __restrict__ w,
               const float* __restrict__ bb, _Float16* __restrict__ xn) {
    __shared__ float tile[DIM_C * 33];   // [c][j], pad 33 -> conflict-free both phases
    __shared__ float ps[256], ps2[256];
    __shared__ float mu_s[32], rs_s[32];
    const int b  = blockIdx.y;
    const int l0 = blockIdx.x * 32;
    const int t  = threadIdx.x;
    const int j  = t & 31, cb = t >> 5;

    float s = 0.f, s2 = 0.f;
    #pragma unroll 4
    for (int i = 0; i < DIM_C / 8; ++i) {      // 48 iters, c = cb + 8*i
        int c = cb + i * 8;
        float v = x[((size_t)b * DIM_C + c) * L_SEQ + l0 + j];  // coalesced over j
        tile[c * 33 + j] = v;
        s += v; s2 += v * v;
    }
    ps[t] = s; ps2[t] = s2;
    __syncthreads();
    if (t < 32) {
        float a = 0.f, a2 = 0.f;
        #pragma unroll
        for (int g = 0; g < 8; ++g) { a += ps[g * 32 + t]; a2 += ps2[g * 32 + t]; }
        float mu  = a * (1.f / DIM_C);
        float var = a2 * (1.f / DIM_C) - mu * mu;
        mu_s[t] = mu;
        rs_s[t] = rsqrtf(var + 1e-5f);
    }
    __syncthreads();
    for (int idx = t; idx < 32 * DIM_C; idx += 256) {
        int jj = idx / DIM_C, c = idx - jj * DIM_C;     // consecutive t -> consecutive c
        float v = (tile[c * 33 + jj] - mu_s[jj]) * rs_s[jj] * w[c] + bb[c];
        xn[((size_t)(b * L_SEQ + l0 + jj)) * DIM_C + c] = (_Float16)v;
    }
}

// ---------------------------------------------------------------------------
// WMMA GEMM:  C[M,N] = A[M,K] (f16, row-major) * W[N,K]^T (f16, row-major).
// Wave computes 16(M) x 32(N) with one A fragment and two accumulators;
// block = 8 waves -> 64x64 tile. K must be a multiple of 32.
// out_mode 0: f32 row-major (ld = N).
// out_mode 1: scatter to (B, N, L_SEQ) with coalesced stores via per-wave
//             LDS transpose (final out_proj, M index = b*L + l).
// ---------------------------------------------------------------------------
__global__ __launch_bounds__(256)
void gemm_wmma_f16(const _Float16* __restrict__ A, const _Float16* __restrict__ W,
                   float* __restrict__ Cout, int M, int N, int K, int out_mode) {
    __shared__ float lds_t[8][16 * 17];
    const int lane = threadIdx.x & 31;
    const int wid  = threadIdx.x >> 5;
    const int half = lane >> 4;
    const int l15  = lane & 15;

    const int m0 = blockIdx.x * 64 + (wid >> 1) * 16;
    const int n0 = blockIdx.y * 64 + (wid & 1) * 32;

    v8f acc0 = {}; v8f acc1 = {};

    // Per-lane fragment base pointers (contiguous k = 16*half + i, same order for A and B).
    const _Float16* pa  = A + (size_t)(m0 + l15) * K + half * 16;
    const _Float16* pb0 = W + (size_t)(n0 + l15) * K + half * 16;
    const _Float16* pb1 = W + (size_t)(n0 + 16 + l15) * K + half * 16;

    for (int k0 = 0; k0 < K; k0 += 32) {
        // Prefetch two k-tiles ahead with high locality -> WGP-scope prefetch
        // (next tile's demand loads are already pipelined by the compiler).
        __builtin_prefetch(pa + k0 + 64, 0, 3);
        __builtin_prefetch(pb0 + k0 + 64, 0, 3);
        __builtin_prefetch(pb1 + k0 + 64, 0, 3);

        v8h alo = *(const v8h*)(pa + k0);        // K 16h+0..16h+7 (per half)
        v8h ahi = *(const v8h*)(pa + k0 + 8);    // K 16h+8..16h+15 (per half)
        v8h b0lo = *(const v8h*)(pb0 + k0);
        v8h b0hi = *(const v8h*)(pb0 + k0 + 8);
        v8h b1lo = *(const v8h*)(pb1 + k0);
        v8h b1hi = *(const v8h*)(pb1 + k0 + 8);

        v16h af, bf0, bf1;
        #pragma unroll
        for (int i = 0; i < 8; ++i) {
            af[i] = alo[i];  af[8 + i] = ahi[i];
            bf0[i] = b0lo[i]; bf0[8 + i] = b0hi[i];
            bf1[i] = b1lo[i]; bf1[8 + i] = b1hi[i];
        }
        acc0 = __builtin_amdgcn_wmma_f32_16x16x32_f16(false, af, false, bf0,
                                                      (short)0, acc0, false, false);
        acc1 = __builtin_amdgcn_wmma_f32_16x16x32_f16(false, af, false, bf1,
                                                      (short)0, acc1, false, false);
    }

    if (out_mode == 0) {
        #pragma unroll
        for (int v = 0; v < 8; ++v) {
            int row = m0 + v + half * 8;
            Cout[(size_t)row * N + n0 + l15]      = acc0[v];
            Cout[(size_t)row * N + n0 + 16 + l15] = acc1[v];
        }
    } else {
        float* t = lds_t[wid];
        // ---- tile 0 (cols n0 .. n0+15) ----
        #pragma unroll
        for (int v = 0; v < 8; ++v) t[(v + half * 8) * 17 + l15] = acc0[v];
        __syncthreads();
        #pragma unroll
        for (int r = 0; r < 8; ++r) {
            int ci = r * 2 + half;
            int mg = m0 + l15;
            int bbatch = mg >> 12, ll = mg & (L_SEQ - 1);
            Cout[((size_t)bbatch * N + n0 + ci) * L_SEQ + ll] = t[l15 * 17 + ci];
        }
        __syncthreads();
        // ---- tile 1 (cols n0+16 .. n0+31) ----
        #pragma unroll
        for (int v = 0; v < 8; ++v) t[(v + half * 8) * 17 + l15] = acc1[v];
        __syncthreads();
        #pragma unroll
        for (int r = 0; r < 8; ++r) {
            int ci = r * 2 + half;
            int mg = m0 + l15;
            int bbatch = mg >> 12, ll = mg & (L_SEQ - 1);
            Cout[((size_t)bbatch * N + n0 + 16 + ci) * L_SEQ + ll] = t[l15 * 17 + ci];
        }
    }
}

// ---------------------------------------------------------------------------
// Depthwise causal conv (D_CONV=4) + bias + SiLU.  xz is (BL, 1536); the xb
// half is its first 768 columns.  One block per token, one thread per channel.
// ---------------------------------------------------------------------------
__global__ __launch_bounds__(768)
void conv_silu_kernel(const float* __restrict__ xz, const float* __restrict__ cw,
                      const float* __restrict__ cb, float* __restrict__ xc) {
    size_t tok = blockIdx.x;
    int l = (int)(tok & (L_SEQ - 1));
    int d = threadIdx.x;
    float acc = cb[d];
    #pragma unroll
    for (int j = 0; j < 4; ++j) {
        int li = l - 3 + j;
        if (li >= 0) acc += cw[d * 4 + j] * xz[(tok - 3 + j) * N_XZ + d];
    }
    float sv = acc / (1.f + __expf(-acc));   // silu
    xc[tok * D_INNER + d] = sv;
}

// ---------------------------------------------------------------------------
// Fused x_proj (56 x 768 GEMV per token) + dt projection (768 x 24) +
// softplus.  One block per token; xc row staged in LDS, the 56 dots split
// 4-ways across 224 threads, then 768 dt outputs over 3 rounds.
// ---------------------------------------------------------------------------
__global__ __launch_bounds__(256)
void xproj_dt_kernel(const float* __restrict__ xc, const float* __restrict__ xpw,
                     const float* __restrict__ dtw, const float* __restrict__ dtb,
                     float* __restrict__ xdbl, float* __restrict__ dtf) {
    __shared__ float row[D_INNER];
    __shared__ float part[224];
    __shared__ float xd[N_XDBL];
    size_t tok = blockIdx.x;
    int t = threadIdx.x;
    for (int i = t; i < D_INNER; i += 256) row[i] = xc[tok * D_INNER + i];
    __syncthreads();
    if (t < 224) {
        int e = t >> 2, q = t & 3;
        const float* w = xpw + (size_t)e * D_INNER + q * 192;
        const float* r = row + q * 192;
        float s = 0.f;
        #pragma unroll 8
        for (int i = 0; i < 192; ++i) s += r[i] * w[i];
        part[t] = s;
    }
    __syncthreads();
    if (t < N_XDBL) {
        float s = part[4 * t] + part[4 * t + 1] + part[4 * t + 2] + part[4 * t + 3];
        xd[t] = s;
        xdbl[tok * N_XDBL + t] = s;
    }
    __syncthreads();
    #pragma unroll
    for (int r3 = 0; r3 < 3; ++r3) {
        int d = t + r3 * 256;
        const float* w = dtw + (size_t)d * DT_RANK;
        float s = dtb[d];
        #pragma unroll
        for (int r = 0; r < DT_RANK; ++r) s += xd[r] * w[r];
        float sp = (s > 20.f) ? s : log1pf(__expf(s));   // softplus
        dtf[tok * D_INNER + d] = sp;
    }
}

// ---------------------------------------------------------------------------
// Selective scan.  Each 16-lane group owns one (b, d) channel; lane = state n.
// h_l = exp(dt*A)*h_{l-1} + dt*x*B_l ; y_l = sum_n h*C_l (shfl_xor reduce).
// grid = (D_INNER/16, B), block = 256 (16 channels / block).
// ---------------------------------------------------------------------------
__global__ __launch_bounds__(256)
void scan_kernel(const float* __restrict__ dtf, const float* __restrict__ xc,
                 const float* __restrict__ xdbl, const float* __restrict__ Alog,
                 float* __restrict__ y) {
    const int b = blockIdx.y;
    const int d = blockIdx.x * 16 + (threadIdx.x >> 4);
    const int n = threadIdx.x & 15;
    const float A = -__expf(Alog[d * D_STATE + n]);
    float h = 0.f;
    const size_t base = (size_t)b * L_SEQ;
    for (int l = 0; l < L_SEQ; ++l) {
        size_t tok = base + l;
        float dt = dtf[tok * D_INNER + d];
        float xv = xc[tok * D_INNER + d];
        float Bn = xdbl[tok * N_XDBL + DT_RANK + n];
        float Cn = xdbl[tok * N_XDBL + DT_RANK + D_STATE + n];
        h = __expf(dt * A) * h + (dt * xv) * Bn;
        float p = h * Cn;
        p += __shfl_xor(p, 8, 32);   // masks < 16: stays inside the state group
        p += __shfl_xor(p, 4, 32);
        p += __shfl_xor(p, 2, 32);
        p += __shfl_xor(p, 1, 32);
        if (n == 0) y[tok * D_INNER + d] = p;
    }
}

// ---------------------------------------------------------------------------
// y2 = (y + xc*D) * silu(z)   (z = second half of xz), output f16 for the
// out_proj WMMA.
// ---------------------------------------------------------------------------
__global__ __launch_bounds__(256)
void gate_kernel(const float* __restrict__ y, const float* __restrict__ xc,
                 const float* __restrict__ xz, const float* __restrict__ Dp,
                 _Float16* __restrict__ y2, int total) {
    int i = blockIdx.x * 256 + threadIdx.x;
    if (i >= total) return;
    int d = i % D_INNER;
    size_t tok = (size_t)i / D_INNER;
    float z = xz[tok * N_XZ + D_INNER + d];
    float s = z / (1.f + __expf(-z));
    y2[i] = (_Float16)((y[i] + xc[i] * Dp[d]) * s);
}

// ---------------------------------------------------------------------------
// Host launcher
// ---------------------------------------------------------------------------
extern "C" void kernel_launch(void* const* d_in, const int* in_sizes, int n_in,
                              void* d_out, int out_size, void* d_ws, size_t ws_size,
                              hipStream_t stream) {
    (void)in_sizes; (void)n_in; (void)out_size; (void)ws_size;
    const float* x          = (const float*)d_in[0];
    const float* ln_w       = (const float*)d_in[1];
    const float* ln_b       = (const float*)d_in[2];
    const float* in_proj_w  = (const float*)d_in[3];
    const float* conv_w     = (const float*)d_in[4];
    const float* conv_b     = (const float*)d_in[5];
    const float* x_proj_w   = (const float*)d_in[6];
    const float* dt_w       = (const float*)d_in[7];
    const float* dt_b       = (const float*)d_in[8];
    const float* A_log      = (const float*)d_in[9];
    const float* Dp         = (const float*)d_in[10];
    const float* out_proj_w = (const float*)d_in[11];
    float* out = (float*)d_out;

    char* ws = (char*)d_ws;
    size_t off = 0;
    auto take = [&](size_t bytes) -> char* {
        char* p = ws + off;
        off = (off + bytes + 255) & ~(size_t)255;
        return p;
    };
    _Float16* xn     = (_Float16*)take((size_t)BL_TOK * DIM_C * 2);
    _Float16* w_in16 = (_Float16*)take((size_t)N_XZ * DIM_C * 2);
    _Float16* w_out16= (_Float16*)take((size_t)DIM_C * D_INNER * 2);
    float*    xz     = (float*)   take((size_t)BL_TOK * N_XZ * 4);
    float*    xc     = (float*)   take((size_t)BL_TOK * D_INNER * 4);
    float*    xdbl   = (float*)   take((size_t)BL_TOK * N_XDBL * 4);
    float*    dtf    = (float*)   take((size_t)BL_TOK * D_INNER * 4);
    float*    yb     = (float*)   take((size_t)BL_TOK * D_INNER * 4);
    _Float16* y2     = (_Float16*)take((size_t)BL_TOK * D_INNER * 2);

    // 1) weight conversion to f16
    {
        int n1 = N_XZ * DIM_C;      // 589824
        int n2 = DIM_C * D_INNER;   // 294912
        cvt_f32_f16<<<(n1 + 255) / 256, 256, 0, stream>>>(in_proj_w, w_in16, n1);
        cvt_f32_f16<<<(n2 + 255) / 256, 256, 0, stream>>>(out_proj_w, w_out16, n2);
    }
    // 2) layernorm -> xn f16 (B*L, 384)
    ln_kernel<<<dim3(L_SEQ / 32, B_SZ), 256, 0, stream>>>(x, ln_w, ln_b, xn);
    // 3) in_proj: xz (BL,1536) = xn (BL,384) x W^T
    gemm_wmma_f16<<<dim3(BL_TOK / 64, N_XZ / 64), 256, 0, stream>>>(
        xn, w_in16, xz, BL_TOK, N_XZ, DIM_C, 0);
    // 4) depthwise conv + silu -> xc (BL,768)
    conv_silu_kernel<<<BL_TOK, D_INNER, 0, stream>>>(xz, conv_w, conv_b, xc);
    // 5) x_proj + dt projection + softplus
    xproj_dt_kernel<<<BL_TOK, 256, 0, stream>>>(xc, x_proj_w, dt_w, dt_b, xdbl, dtf);
    // 6) selective scan -> yb (BL,768)
    scan_kernel<<<dim3(D_INNER / 16, B_SZ), 256, 0, stream>>>(dtf, xc, xdbl, A_log, yb);
    // 7) gate -> y2 f16
    {
        int total = BL_TOK * D_INNER;
        gate_kernel<<<(total + 255) / 256, 256, 0, stream>>>(yb, xc, xz, Dp, y2, total);
    }
    // 8) out_proj with transposed (B, C, L) store
    gemm_wmma_f16<<<dim3(BL_TOK / 64, DIM_C / 64), 256, 0, stream>>>(
        y2, w_out16, out, BL_TOK, DIM_C, D_INNER, 1);
}
